// WGCN_layer_41566693491077
// MI455X (gfx1250) — matrix operand order repaired and checked by reading, not compiled, not run on hardware
//
#include <hip/hip_runtime.h>
#include <hip/hip_bf16.h>
#include <math.h>

typedef __attribute__((ext_vector_type(16))) _Float16 v16h;
typedef __attribute__((ext_vector_type(8)))  _Float16 v8h;
typedef __attribute__((ext_vector_type(8)))  float    v8f;

#define DSUP 128          // distribution support size D
#define KNBR 16           // padded neighbors (excl self)
#define KN   17           // neighbors + self
#define MPAD 32           // KN padded to 2 WMMA M-tiles
#define EPS_INV 10.0f     // 1/0.1
#define TINYF 1e-30f
#define SINK_ITERS 20

// ---- WMMA fragment loaders (contiguous k order, same for A and B, wave32) -------------------
// A (16x32 f16): lane L<16 holds row L, halves0-15 = K {0..15};
// lane L>=16 holds row L-16 with K {16..31}.
__device__ __forceinline__ v16h load_A_frag(const _Float16* base, int mrow0, int ks, int lane) {
  int r = mrow0 + (lane & 15);
  int khi = (lane >> 4) * 16;                      // 0 or 16
  const _Float16* p = base + r * DSUP + ks * 32 + khi;
  v8h lo = *(const v8h*)(p);                       // K = ks*32+khi .. +7
  v8h hi = *(const v8h*)(p + 8);                   // K = ks*32+khi+8 .. +15
  v16h a;
#pragma unroll
  for (int i = 0; i < 8; ++i) { a[i] = lo[i]; a[8 + i] = hi[i]; }
  return a;
}

// B (32x16 f16): lane holds column N = lane%16; halves i = K = ks*32 + (lane/16)*16 + i.
// B = K (row-major, rows = contraction dim) -> strided reads.
__device__ __forceinline__ v16h load_B_K(const _Float16* sK_, int ntile, int ks, int lane) {
  int col = ntile * 16 + (lane & 15);
  int kb  = ks * 32 + ((lane >> 4) << 4);
  v16h b;
#pragma unroll
  for (int i = 0; i < 16; ++i) b[i] = sK_[(kb + i) * DSUP + col];
  return b;
}

// B = K^T: element (K=kb+i, N=col) of K^T is K[col][kb+i] -> contiguous reads.
__device__ __forceinline__ v16h load_B_KT(const _Float16* sK_, int ntile, int ks, int lane) {
  int col = ntile * 16 + (lane & 15);
  int kb  = ks * 32 + ((lane >> 4) << 4);
  const _Float16* p = sK_ + col * DSUP + kb;
  v8h lo = *(const v8h*)(p);
  v8h hi = *(const v8h*)(p + 8);
  v16h b;
#pragma unroll
  for (int i = 0; i < 8; ++i) { b[i] = lo[i]; b[8 + i] = hi[i]; }
  return b;
}

// ---- One barycenter hop: one workgroup per node, Sinkhorn loop in LDS -----
__global__ __launch_bounds__(256) void wgcn_bary_hop(
    const float* __restrict__ X_in, const float* __restrict__ Cmat,
    const int* __restrict__ nidx, float* __restrict__ X_out) {

  __shared__ __align__(16) _Float16 sK [DSUP * DSUP];  // Gibbs kernel, f16   (32 KB)
  __shared__ __align__(16) _Float16 sMu[KN * DSUP];    // normalized inputs   (4.25 KB)
  __shared__ __align__(16) _Float16 sU [MPAD * DSUP];  // u (pad rows = 0)    (8 KB)
  __shared__ __align__(16) _Float16 sV [MPAD * DSUP];  // v (pad rows = 0)    (8 KB)
  __shared__ float sLog[DSUP];                         // sum_k log2(Kv)      (0.5 KB)
  __shared__ float sB [DSUP];                          // barycenter b        (0.5 KB)

  const int n    = blockIdx.x;
  const int tid  = threadIdx.x;
  const int lane = tid & 31;
  const int w    = tid >> 5;      // 8 waves

  // Gibbs kernel K = exp(-C/eps) in f16 (K is tiny and L2-resident globally)
  for (int i = tid; i < DSUP * DSUP; i += 256)
    sK[i] = (_Float16)__expf(-EPS_INV * Cmat[i]);

  // init u = 1 on valid rows, 0 on pad rows (pad rows never rewritten)
  for (int i = tid; i < MPAD * DSUP; i += 256)
    sU[i] = (_Float16)((i < KN * DSUP) ? 1.0f : 0.0f);
  if (tid < DSUP) sLog[tid] = 0.0f;

  // gather neighbor distributions (+ self) and row-normalize; one row per wave
  for (int r = w; r < KN; r += 8) {
    int node = (r < KNBR) ? nidx[n * KNBR + r] : n;
    const float4* src = (const float4*)(X_in + (long)node * DSUP);
    float4 x = src[lane];                       // 4 elems per lane
    float s = x.x + x.y + x.z + x.w;
#pragma unroll
    for (int off = 16; off > 0; off >>= 1)
      s += __shfl_xor(s, off, 32);
    float inv = __builtin_amdgcn_rcpf(s + TINYF);
    _Float16* dst = sMu + r * DSUP + lane * 4;
    dst[0] = (_Float16)(x.x * inv);
    dst[1] = (_Float16)(x.y * inv);
    dst[2] = (_Float16)(x.z * inv);
    dst[3] = (_Float16)(x.w * inv);
  }
  __syncthreads();

  // Wave w owns output tiles t = 2w, 2w+1. mtile = w>>2 is WAVE-UNIFORM.
  const int mtile = w >> 2;
  const int mbase = mtile * 16 + ((lane >> 4) << 3);   // C-layout row base
  int col[2];
  v16h Bk[2][4], Bt[2][4];
  float muR[2][8];                                     // mu in C layout (loop-invariant)
#pragma unroll
  for (int j = 0; j < 2; ++j) {
    int ntile = ((w & 3) << 1) | j;
    col[j] = ntile * 16 + (lane & 15);
#pragma unroll
    for (int ks = 0; ks < 4; ++ks) {
      Bk[j][ks] = load_B_K (sK, ntile, ks, lane);
      Bt[j][ks] = load_B_KT(sK, ntile, ks, lane);
    }
#pragma unroll
    for (int p = 0; p < 8; ++p) {
      int row = mbase + p;
      muR[j][p] = (row < KN) ? (float)sMu[row * DSUP + col[j]] : 0.0f;
    }
  }

  const float wbar = 1.0f / (float)KN;

#pragma unroll 1
  for (int it = 0; it < SINK_ITERS; ++it) {
    float c2[2][8];                                    // Kv kept in registers

    // ---- KTu = u * K ; v = mu * rcp(KTu + tiny) (branchless, pad mu = 0) --
#pragma unroll
    for (int j = 0; j < 2; ++j) {
      v8f c = {};
#pragma unroll
      for (int ks = 0; ks < 4; ++ks) {
        v16h a = load_A_frag(sU, mtile * 16, ks, lane);
        c = __builtin_amdgcn_wmma_f32_16x16x32_f16(false, a, false, Bk[j][ks],
                                                   (short)0, c, false, false);
      }
#pragma unroll
      for (int p = 0; p < 8; ++p)
        sV[(mbase + p) * DSUP + col[j]] =
            (_Float16)(muR[j][p] * __builtin_amdgcn_rcpf(c[p] + TINYF));
    }
    __syncthreads();

    // ---- Kv = v * K^T ; accumulate sum_k log2(Kv) (one atomic per lane) ---
#pragma unroll
    for (int j = 0; j < 2; ++j) {
      v8f c = {};
#pragma unroll
      for (int ks = 0; ks < 4; ++ks) {
        v16h a = load_A_frag(sV, mtile * 16, ks, lane);
        c = __builtin_amdgcn_wmma_f32_16x16x32_f16(false, a, false, Bt[j][ks],
                                                   (short)0, c, false, false);
      }
#pragma unroll
      for (int p = 0; p < 8; ++p) c2[j][p] = c[p] + TINYF;

      if (mtile == 0) {                                // rows 0..15: all valid
        float ls = 0.0f;
#pragma unroll
        for (int p = 0; p < 8; ++p) ls += __builtin_amdgcn_logf(c2[j][p]);
        atomicAdd(&sLog[col[j]], ls);                  // ds_add_f32
      } else if (lane < 16) {                          // only row 16 valid
        atomicAdd(&sLog[col[j]], __builtin_amdgcn_logf(c2[j][0]));
      }
    }
    __syncthreads();

    // ---- b = 2^(mean_k log2(Kv)) ; reset accumulator ----
    if (tid < DSUP) {
      sB[tid]   = __builtin_amdgcn_exp2f(sLog[tid] * wbar);
      sLog[tid] = 0.0f;
    }
    __syncthreads();

    // ---- u = b * rcp(Kv) straight from accumulator registers ----
#pragma unroll
    for (int j = 0; j < 2; ++j) {
      float bb = sB[col[j]];
      if (mtile == 0) {
#pragma unroll
        for (int p = 0; p < 8; ++p)
          sU[(mbase + p) * DSUP + col[j]] =
              (_Float16)(bb * __builtin_amdgcn_rcpf(c2[j][p]));
      } else if (lane < 16) {
        sU[16 * DSUP + col[j]] =
            (_Float16)(bb * __builtin_amdgcn_rcpf(c2[j][0]));
      }
    }
    __syncthreads();
  }

  if (tid < DSUP) X_out[(long)n * DSUP + tid] = sB[tid];
}

extern "C" void kernel_launch(void* const* d_in, const int* in_sizes, int n_in,
                              void* d_out, int out_size, void* d_ws, size_t ws_size,
                              hipStream_t stream) {
  const float* trans_X    = (const float*)d_in[0];
  const float* costMatrix = (const float*)d_in[1];
  const int*   neigh_idx  = (const int*)d_in[2];
  float* out = (float*)d_out;
  float* X1  = (float*)d_ws;                 // hop-1 barycenters [N, D]

  const int n_nodes = in_sizes[0] / DSUP;    // 20000
  dim3 grid(n_nodes), block(256);

  // H = 2 barycenter hops
  hipLaunchKernelGGL(wgcn_bary_hop, grid, block, 0, stream,
                     trans_X, costMatrix, neigh_idx, X1);
  hipLaunchKernelGGL(wgcn_bary_hop, grid, block, 0, stream,
                     X1, costMatrix, neigh_idx, out);
}
